// Block_21440476741645
// MI455X (gfx1250) — matrix operand order repaired and checked by reading, not compiled, not run on hardware
//
#include <hip/hip_runtime.h>
#include <math.h>

typedef __attribute__((ext_vector_type(16))) _Float16 v16h;
typedef __attribute__((ext_vector_type(8)))  _Float16 v8h;
typedef __attribute__((ext_vector_type(8)))  float    v8f;

#define N_EMBD 768
#define N_HEAD 12
#define HS     64
#define TSEQ   1024
#define BATCH  8
#define MROWS  (BATCH * TSEQ)   // 8192
#define FFN    (4 * N_EMBD)     // 3072
#define INF_F  __builtin_inff()

static __device__ inline v8f wmma_f16(v16h a, v16h b, v8f c) {
  return __builtin_amdgcn_wmma_f32_16x16x32_f16(false, a, false, b, (short)0, c, false, false);
}

// A fragment (16x32 f16): elems 0..7 -> K = kk + half*16 + i,
//                         elems 8..15 -> K = kk + half*16 + 8 + i
static __device__ inline v16h load_afrag(const _Float16* __restrict__ arow,
                                         int kk, int half) {
  v8h lo = *(const v8h*)(arow + kk + half * 16);
  v8h hi = *(const v8h*)(arow + kk + half * 16 + 8);
  return __builtin_shufflevector(lo, hi, 0, 1, 2, 3, 4, 5, 6, 7,
                                 8, 9, 10, 11, 12, 13, 14, 15);
}

// B fragment (32x16 f16) from transposed weight Wt[N,K]: per lane the column
// is fixed and the 16 K-halves are contiguous -> one 32-byte load.
static __device__ inline v16h load_bfrag(const _Float16* __restrict__ Bt,
                                         size_t n, int K, int kk, int half) {
  return *(const v16h*)(Bt + n * (size_t)K + kk + half * 16);
}

// ---------------------------------------------------------------------------
// fp32 weight [K,N] -> f16 transposed [N,K].
// ---------------------------------------------------------------------------
__global__ void transpose_to_f16(const float* __restrict__ W,
                                 _Float16* __restrict__ Wt, int K, int N) {
  int idx = blockIdx.x * blockDim.x + threadIdx.x;
  if (idx >= K * N) return;
  int k = idx / N, n = idx % N;
  Wt[(size_t)n * K + k] = (_Float16)W[idx];
}

// ---------------------------------------------------------------------------
// Row LayerNorm over C=768 -> f16 activations. One block (256 thr) per row.
// ---------------------------------------------------------------------------
__global__ __launch_bounds__(256)
void layernorm_f16(const float* __restrict__ x, const float* __restrict__ g,
                   const float* __restrict__ b, _Float16* __restrict__ out) {
  __shared__ float red[256];
  const int row = blockIdx.x, tid = threadIdx.x;
  const float* xr = x + (size_t)row * N_EMBD;
  float v0 = xr[tid], v1 = xr[tid + 256], v2 = xr[tid + 512];

  red[tid] = v0 + v1 + v2;
  __syncthreads();
  for (int off = 128; off > 0; off >>= 1) {
    if (tid < off) red[tid] += red[tid + off];
    __syncthreads();
  }
  float mean = red[0] * (1.0f / N_EMBD);
  __syncthreads();

  red[tid] = v0 * v0 + v1 * v1 + v2 * v2;
  __syncthreads();
  for (int off = 128; off > 0; off >>= 1) {
    if (tid < off) red[tid] += red[tid + off];
    __syncthreads();
  }
  float var = red[0] * (1.0f / N_EMBD) - mean * mean;
  float rstd = rsqrtf(var + 1e-5f);

  _Float16* orow = out + (size_t)row * N_EMBD;
  orow[tid]       = (_Float16)((v0 - mean) * rstd * g[tid]       + b[tid]);
  orow[tid + 256] = (_Float16)((v1 - mean) * rstd * g[tid + 256] + b[tid + 256]);
  orow[tid + 512] = (_Float16)((v2 - mean) * rstd * g[tid + 512] + b[tid + 512]);
}

// ---------------------------------------------------------------------------
// WMMA GEMM: Out[M,N] = A[M,K](f16) x W[K,N] (given as Wt[N,K] f16).
// Block = 128 threads = 4 waves; each wave owns a 32(M) x 64(N) tile:
// 2 A-fragments x 4 B-fragments = 8 WMMA per 32-K step (12 b128 loads),
// software-pipelined with register double buffering. Grid = (M/128, N/64).
// MODE 0: outH = acc (f16)                [QKV]
// MODE 1: outF = resid + acc + bias (f32) [Wo, W2 residual adds]
// MODE 2: outH = gelu(acc + bias) (f16)   [W1]
// ---------------------------------------------------------------------------
template <int MODE>
__global__ __launch_bounds__(128)
void gemm_wmma(const _Float16* __restrict__ A, const _Float16* __restrict__ Bt,
               const float* __restrict__ bias, const float* __restrict__ resid,
               float* __restrict__ outF, _Float16* __restrict__ outH,
               int M, int N, int K) {
  const int lane = threadIdx.x & 31;
  const int wave = threadIdx.x >> 5;
  const int half = lane >> 4;
  const int col  = lane & 15;
  const int m0 = blockIdx.x * 128 + wave * 32;
  const int n0 = blockIdx.y * 64;

  v8f acc0[4], acc1[4];
#pragma unroll
  for (int t = 0; t < 4; ++t) { acc0[t] = (v8f){}; acc1[t] = (v8f){}; }

  const _Float16* arow0 = A + (size_t)(m0 + col) * K;
  const _Float16* arow1 = A + (size_t)(m0 + 16 + col) * K;

  // current fragments
  v16h a0 = load_afrag(arow0, 0, half);
  v16h a1 = load_afrag(arow1, 0, half);
  v16h b0 = load_bfrag(Bt, n0 + 0 + col, K, 0, half);
  v16h b1 = load_bfrag(Bt, n0 + 16 + col, K, 0, half);
  v16h b2 = load_bfrag(Bt, n0 + 32 + col, K, 0, half);
  v16h b3 = load_bfrag(Bt, n0 + 48 + col, K, 0, half);

  int kk = 0;
  for (; kk + 32 < K; kk += 32) {
    // prefetch next K-step while WMMAs consume the current fragments
    int kn = kk + 32;
    v16h c0 = load_afrag(arow0, kn, half);
    v16h c1 = load_afrag(arow1, kn, half);
    v16h d0 = load_bfrag(Bt, n0 + 0 + col, K, kn, half);
    v16h d1 = load_bfrag(Bt, n0 + 16 + col, K, kn, half);
    v16h d2 = load_bfrag(Bt, n0 + 32 + col, K, kn, half);
    v16h d3 = load_bfrag(Bt, n0 + 48 + col, K, kn, half);

    acc0[0] = wmma_f16(a0, b0, acc0[0]);
    acc0[1] = wmma_f16(a0, b1, acc0[1]);
    acc0[2] = wmma_f16(a0, b2, acc0[2]);
    acc0[3] = wmma_f16(a0, b3, acc0[3]);
    acc1[0] = wmma_f16(a1, b0, acc1[0]);
    acc1[1] = wmma_f16(a1, b1, acc1[1]);
    acc1[2] = wmma_f16(a1, b2, acc1[2]);
    acc1[3] = wmma_f16(a1, b3, acc1[3]);

    a0 = c0; a1 = c1; b0 = d0; b1 = d1; b2 = d2; b3 = d3;
  }
  // epilogue K-step
  acc0[0] = wmma_f16(a0, b0, acc0[0]);
  acc0[1] = wmma_f16(a0, b1, acc0[1]);
  acc0[2] = wmma_f16(a0, b2, acc0[2]);
  acc0[3] = wmma_f16(a0, b3, acc0[3]);
  acc1[0] = wmma_f16(a1, b0, acc1[0]);
  acc1[1] = wmma_f16(a1, b1, acc1[1]);
  acc1[2] = wmma_f16(a1, b2, acc1[2]);
  acc1[3] = wmma_f16(a1, b3, acc1[3]);

#pragma unroll
  for (int g = 0; g < 2; ++g) {
#pragma unroll
    for (int t = 0; t < 4; ++t) {
#pragma unroll
      for (int r = 0; r < 8; ++r) {
        int row = m0 + g * 16 + r + 8 * half;
        int c   = n0 + t * 16 + col;
        size_t idx = (size_t)row * N + c;
        float v = g ? acc1[t][r] : acc0[t][r];
        if (MODE == 0) {
          outH[idx] = (_Float16)v;
        } else if (MODE == 1) {
          outF[idx] = resid[idx] + v + bias[c];
        } else {  // exact GELU
          v += bias[c];
          float ge = 0.5f * v * (1.0f + erff(v * 0.70710678118f));
          outH[idx] = (_Float16)ge;
        }
      }
    }
  }
}

// ---------------------------------------------------------------------------
// Flash-attention: one wave per (b, h, 16-row block). Per 32-key block:
//  - V tile staged to LDS with GLOBAL_LOAD_ASYNC_TO_LDS_B128 (ASYNCcnt),
//    overlapped with S = Q@K^T (4 WMMA) and the online softmax,
//  - s_wait_asynccnt 0 before O += P@V (4 WMMA).
// ---------------------------------------------------------------------------
__global__ __launch_bounds__(32)
void attn_kernel(const _Float16* __restrict__ Q, const _Float16* __restrict__ Km,
                 const _Float16* __restrict__ Vm, _Float16* __restrict__ O,
                 float scale) {
  __shared__ alignas(32) _Float16 Pbuf[16 * 32];
  __shared__ alignas(32) _Float16 Vbuf[32 * 64];

  const int lane = threadIdx.x;
  const int half = lane >> 4;
  const int col  = lane & 15;
  const int m0   = blockIdx.x * 16;
  const int bh   = blockIdx.y;
  const size_t base =
      (size_t)(bh / N_HEAD) * TSEQ * N_EMBD + (size_t)(bh % N_HEAD) * HS;

  // Q A-fragments for K-dim chunks [0,32) and [32,64), kept in registers.
  const _Float16* qp = Q + base + (size_t)(m0 + col) * N_EMBD;
  v16h qa0, qa1;
  {
    v8h lo = *(const v8h*)(qp + half * 16);
    v8h hi = *(const v8h*)(qp + half * 16 + 8);
    qa0 = __builtin_shufflevector(lo, hi, 0, 1, 2, 3, 4, 5, 6, 7,
                                  8, 9, 10, 11, 12, 13, 14, 15);
    lo = *(const v8h*)(qp + 32 + half * 16);
    hi = *(const v8h*)(qp + 40 + half * 16);
    qa1 = __builtin_shufflevector(lo, hi, 0, 1, 2, 3, 4, 5, 6, 7,
                                  8, 9, 10, 11, 12, 13, 14, 15);
  }

  float mrow[8], lrow[8];
  v8f oacc[4];
#pragma unroll
  for (int r = 0; r < 8; ++r) { mrow[r] = -INF_F; lrow[r] = 0.0f; }
#pragma unroll
  for (int t = 0; t < 4; ++t) oacc[t] = (v8f){};

  for (int j0 = 0; j0 <= m0 + 15; j0 += 32) {
    // ---- async-stage V block (32 keys x 64 halves = 4KB) into LDS --------
    // 8 chunks per lane, 16B each; tracked by ASYNCcnt, overlapped with S.
#pragma unroll
    for (int it = 0; it < 8; ++it) {
      int c = lane + it * 32;
      int key = c >> 3, ch = c & 7;
      unsigned ldsoff =
          (unsigned)(unsigned long long)(uintptr_t)&Vbuf[key * 64 + ch * 8];
      unsigned long long gaddr = (unsigned long long)(uintptr_t)(
          Vm + base + (size_t)(j0 + key) * N_EMBD + ch * 8);
      asm volatile("global_load_async_to_lds_b128 %0, %1, off"
                   :: "v"(ldsoff), "v"(gaddr)
                   : "memory");
    }

    // ---- S = Q @ K^T for keys [j0, j0+32): distinct fragment registers so
    // the 8 b128 loads clause together ahead of the 4 WMMAs. -------------
    const _Float16* kp0 = Km + base + (size_t)(j0 + col) * N_EMBD;
    const _Float16* kp1 = Km + base + (size_t)(j0 + 16 + col) * N_EMBD;
    v16h kb00 = *(const v16h*)(kp0 + half * 16);
    v16h kb01 = *(const v16h*)(kp1 + half * 16);
    v16h kb10 = *(const v16h*)(kp0 + 32 + half * 16);
    v16h kb11 = *(const v16h*)(kp1 + 32 + half * 16);
    v8f s0 = (v8f){}, s1 = (v8f){};
    s0 = wmma_f16(qa0, kb00, s0);
    s1 = wmma_f16(qa0, kb01, s1);
    s0 = wmma_f16(qa1, kb10, s0);
    s1 = wmma_f16(qa1, kb11, s1);

    // ---- online softmax: rows (r + 8*half) across 16-lane groups ---------
    float alpha[8];
#pragma unroll
    for (int r = 0; r < 8; ++r) {
      int grow = m0 + r + 8 * half;
      float e0 = s0[r] * scale;
      float e1 = s1[r] * scale;
      if (j0 + col > grow) e0 = -INF_F;        // causal mask
      if (j0 + 16 + col > grow) e1 = -INF_F;
      float mx = fmaxf(e0, e1);
      mx = fmaxf(mx, __shfl_xor(mx, 1, 32));
      mx = fmaxf(mx, __shfl_xor(mx, 2, 32));
      mx = fmaxf(mx, __shfl_xor(mx, 4, 32));
      mx = fmaxf(mx, __shfl_xor(mx, 8, 32));
      float mn = fmaxf(mrow[r], mx);
      float al = __expf(mrow[r] - mn);
      float p0 = __expf(e0 - mn);
      float p1 = __expf(e1 - mn);
      float ps = p0 + p1;
      ps += __shfl_xor(ps, 1, 32);
      ps += __shfl_xor(ps, 2, 32);
      ps += __shfl_xor(ps, 4, 32);
      ps += __shfl_xor(ps, 8, 32);
      lrow[r] = lrow[r] * al + ps;
      mrow[r] = mn;
      alpha[r] = al;
      Pbuf[(r + 8 * half) * 32 + col]      = (_Float16)p0;
      Pbuf[(r + 8 * half) * 32 + 16 + col] = (_Float16)p1;
    }
#pragma unroll
    for (int t = 0; t < 4; ++t)
#pragma unroll
      for (int r = 0; r < 8; ++r) oacc[t][r] *= alpha[r];

    // P A-fragment (16x32) from LDS (C-layout -> A-layout transpose).
    v16h pa;
    {
      const _Float16* pp = &Pbuf[col * 32];
      v8h lo = *(const v8h*)(pp + half * 16);
      v8h hi = *(const v8h*)(pp + half * 16 + 8);
      pa = __builtin_shufflevector(lo, hi, 0, 1, 2, 3, 4, 5, 6, 7,
                                   8, 9, 10, 11, 12, 13, 14, 15);
    }

    // Async V copy must have landed before the P@V WMMAs consume Vbuf.
    asm volatile("s_wait_asynccnt 0x0" ::: "memory");

    // O += P @ V : 4 WMMA over the 64 hs columns.
#pragma unroll
    for (int t = 0; t < 4; ++t) {
      v16h bv;
#pragma unroll
      for (int j = 0; j < 16; ++j)
        bv[j] = Vbuf[(half * 16 + j) * 64 + t * 16 + col];
      oacc[t] = wmma_f16(pa, bv, oacc[t]);
    }
  }

  // O / l, store f16 for the Wo GEMM.
#pragma unroll
  for (int t = 0; t < 4; ++t)
#pragma unroll
    for (int r = 0; r < 8; ++r) {
      float val = oacc[t][r] / lrow[r];
      O[base + (size_t)(m0 + r + 8 * half) * N_EMBD + t * 16 + col] =
          (_Float16)val;
    }
}

// ---------------------------------------------------------------------------
extern "C" void kernel_launch(void* const* d_in, const int* in_sizes, int n_in,
                              void* d_out, int out_size, void* d_ws,
                              size_t ws_size, hipStream_t stream) {
  const float* x   = (const float*)d_in[0];
  const float* Wq  = (const float*)d_in[1];
  const float* Wk  = (const float*)d_in[2];
  const float* Wv  = (const float*)d_in[3];
  const float* Wo  = (const float*)d_in[4];
  const float* bo  = (const float*)d_in[5];
  const float* W1  = (const float*)d_in[6];
  const float* b1  = (const float*)d_in[7];
  const float* W2  = (const float*)d_in[8];
  const float* b2  = (const float*)d_in[9];
  const float* g1  = (const float*)d_in[10];
  const float* be1 = (const float*)d_in[11];
  const float* g2  = (const float*)d_in[12];
  const float* be2 = (const float*)d_in[13];
  float* out = (float*)d_out;

  char* ws = (char*)d_ws;
  const size_t sz16 = (size_t)MROWS * N_EMBD * sizeof(_Float16);  // 12.58 MB
  // q/k/v/o16 region (4*sz16) is reused as the FFN intermediate a1
  _Float16* q16 = (_Float16*)(ws + 0 * sz16);
  _Float16* k16 = (_Float16*)(ws + 1 * sz16);
  _Float16* v16 = (_Float16*)(ws + 2 * sz16);
  _Float16* o16 = (_Float16*)(ws + 3 * sz16);
  _Float16* a1  = (_Float16*)(ws + 0 * sz16);  // alias, used after q/k/v/o dead
  _Float16* h16 = (_Float16*)(ws + 4 * sz16);  // LN1 output, reused for LN2
  float*    x2  = (float*)(ws + 5 * sz16);     // f32 residual stream after attn
  size_t off = 5 * sz16 + (size_t)MROWS * N_EMBD * sizeof(float);
  _Float16* wqt = (_Float16*)(ws + off); off += (size_t)N_EMBD * N_EMBD * 2;
  _Float16* wkt = (_Float16*)(ws + off); off += (size_t)N_EMBD * N_EMBD * 2;
  _Float16* wvt = (_Float16*)(ws + off); off += (size_t)N_EMBD * N_EMBD * 2;
  _Float16* wot = (_Float16*)(ws + off); off += (size_t)N_EMBD * N_EMBD * 2;
  _Float16* w1t = (_Float16*)(ws + off); off += (size_t)N_EMBD * FFN * 2;
  _Float16* w2t = (_Float16*)(ws + off); off += (size_t)FFN * N_EMBD * 2;

  // Weight conversion (fp32 [K,N] -> f16 [N,K]); f16 weights stay L2-resident.
  int ncc = N_EMBD * N_EMBD, ncf = N_EMBD * FFN;
  transpose_to_f16<<<(ncc + 255) / 256, 256, 0, stream>>>(Wq, wqt, N_EMBD, N_EMBD);
  transpose_to_f16<<<(ncc + 255) / 256, 256, 0, stream>>>(Wk, wkt, N_EMBD, N_EMBD);
  transpose_to_f16<<<(ncc + 255) / 256, 256, 0, stream>>>(Wv, wvt, N_EMBD, N_EMBD);
  transpose_to_f16<<<(ncc + 255) / 256, 256, 0, stream>>>(Wo, wot, N_EMBD, N_EMBD);
  transpose_to_f16<<<(ncf + 255) / 256, 256, 0, stream>>>(W1, w1t, N_EMBD, FFN);
  transpose_to_f16<<<(ncf + 255) / 256, 256, 0, stream>>>(W2, w2t, FFN, N_EMBD);

  // LN1 -> h
  layernorm_f16<<<MROWS, 256, 0, stream>>>(x, g1, be1, h16);

  // QKV projections
  dim3 gcc(MROWS / 128, N_EMBD / 64);
  gemm_wmma<0><<<gcc, 128, 0, stream>>>(h16, wqt, nullptr, nullptr, nullptr, q16,
                                        MROWS, N_EMBD, N_EMBD);
  gemm_wmma<0><<<gcc, 128, 0, stream>>>(h16, wkt, nullptr, nullptr, nullptr, k16,
                                        MROWS, N_EMBD, N_EMBD);
  gemm_wmma<0><<<gcc, 128, 0, stream>>>(h16, wvt, nullptr, nullptr, nullptr, v16,
                                        MROWS, N_EMBD, N_EMBD);

  // Flash attention (reference scales by C^-0.5, not hs^-0.5)
  dim3 ga(TSEQ / 16, BATCH * N_HEAD);
  attn_kernel<<<ga, 32, 0, stream>>>(q16, k16, v16, o16,
                                     1.0f / sqrtf((float)N_EMBD));

  // x2 = x + o @ Wo + bo
  gemm_wmma<1><<<gcc, 128, 0, stream>>>(o16, wot, bo, x, x2, nullptr,
                                        MROWS, N_EMBD, N_EMBD);

  // LN2 -> h2 (reuses h16)
  layernorm_f16<<<MROWS, 256, 0, stream>>>(x2, g2, be2, h16);

  // a1 = gelu(h2 @ W1 + b1)   (a1 aliases q/k/v/o region, now dead)
  dim3 gcf(MROWS / 128, FFN / 64);
  gemm_wmma<2><<<gcf, 128, 0, stream>>>(h16, w1t, b1, nullptr, nullptr, a1,
                                        MROWS, FFN, N_EMBD);

  // out = x2 + a1 @ W2 + b2
  gemm_wmma<1><<<gcc, 128, 0, stream>>>(a1, w2t, b2, x2, out, nullptr,
                                        MROWS, N_EMBD, FFN);
}
